// MultiHeadSelfAttention_81673098101123
// MI455X (gfx1250) — matrix operand order repaired and checked by reading, not compiled, not run on hardware
//
#include <hip/hip_runtime.h>

// ---------------------------------------------------------------------------
// CDNA5 (gfx1250) bf16 WMMA multi-head self-attention.
//   B=4, S=2048, D=1024, H=16, hd=64.
// Pipeline (workspace, 88 MB used):
//   [0]  cvt   : x f32 -> xb bf16                       (16 MB)
//   [1]  cvtT  : W* f32 -> WT* bf16 transposed [N][K]   (4 x 2 MB)
//   [2]  gemm  : Q = xb@Wq  -> [B,H,S,hd] bf16          (16 MB)
//               K = xb@Wk  -> [B,H,S,hd] bf16          (16 MB)
//               V = xb@Wv  -> [B,H,hd,S] bf16 (transp) (16 MB)
//   [3]  attn  : flash attention, async global->LDS K/V staging (ASYNCcnt)
//               -> ctx bf16 [B*S, D]                   (16 MB)
//   [4]  gemm  : out = ctx@Wo + bo -> f32 d_out
// ---------------------------------------------------------------------------

#define DEV static __device__ __forceinline__

typedef __attribute__((ext_vector_type(16))) __bf16 v16bf;
typedef __attribute__((ext_vector_type(8)))  float  v8f;

union FragBF {
  v16bf v;
  unsigned int u[8];
};

DEV unsigned short f2bf(float f) {
  unsigned int u = __float_as_uint(f);
  u += 0x7FFFu + ((u >> 16) & 1u);      // round-to-nearest-even
  return (unsigned short)(u >> 16);
}

// A fragment, 16x32 (MxK) bf16; Atile -> (row 0, col 0) of M-tile, K contiguous.
// ISA: lanes 0-15 -> M=lane, K {kbase+0..7, +16..23}; lanes 16-31 -> K {+8..15, +24..31}
DEV v16bf load_a_frag(const unsigned short* __restrict__ Atile, int lda,
                      int kbase, int lane) {
  const int m    = lane & 15;
  const int half = (lane >> 4) & 1;
  const unsigned short* p = Atile + (size_t)m * lda + kbase + 8 * half;
  FragBF f;
#pragma unroll
  for (int r = 0; r < 4; ++r) {
    f.u[r]     = *(const unsigned int*)(p + 2 * r);
    f.u[4 + r] = *(const unsigned int*)(p + 16 + 2 * r);
  }
  return f.v;
}

// B fragment, 32x16 (KxN) bf16 from K-contiguous transposed storage BT[n][k].
// lane(0-15)=N, lane half selects K {0..7,16..23} / {8..15,24..31}; each VGPR = contiguous K pair.
DEV v16bf load_b_frag(const unsigned short* __restrict__ Btile, int ldb,
                      int kbase, int lane) {
  const int n     = lane & 15;
  const int khalf = (lane >> 4) & 1;
  const unsigned short* p = Btile + (size_t)n * ldb + kbase + 8 * khalf;
  FragBF f;
#pragma unroll
  for (int r = 0; r < 8; ++r)
    f.u[r] = *(const unsigned int*)(p + 2 * r + (r >= 4 ? 8 : 0));
  return f.v;
}

DEV v8f wmma_bf16(v16bf a, v16bf b, v8f c) {
  return __builtin_amdgcn_wmma_f32_16x16x32_bf16(false, a, false, b,
                                                 (short)0, c, false, false);
}

// Async global -> LDS copy, 16 bytes per lane (ISA GLOBAL_LOAD_ASYNC_TO_LDS_B128,
// tracked by ASYNCcnt). ldsOff = per-lane LDS byte address, gaddr = per-lane VA.
DEV void async_ld_b128(unsigned ldsOff, const void* gaddr) {
  asm volatile("global_load_async_to_lds_b128 %0, %1, off"
               :: "v"(ldsOff), "v"(gaddr) : "memory");
}
#define WAIT_ASYNCCNT(imm) asm volatile("s_wait_asynccnt " #imm ::: "memory")

// ---------------------------------------------------------------------------
__global__ __launch_bounds__(256)
void cvt_f32_bf16(const float* __restrict__ in, unsigned short* __restrict__ out,
                  size_t n) {
  size_t i = (size_t)blockIdx.x * 256 + threadIdx.x;
  if (i < n) out[i] = f2bf(in[i]);
}

// fp32 W[k][n] -> bf16 WT[n][k], 32x32 LDS tile transpose
__global__ __launch_bounds__(256)
void cvt_transpose_bf16(const float* __restrict__ W,
                        unsigned short* __restrict__ WT, int Dd) {
  __shared__ float tile[32][33];
  const int bx = blockIdx.x * 32;   // n base
  const int by = blockIdx.y * 32;   // k base
  const int tx = threadIdx.x & 31;
  const int ty = threadIdx.x >> 5;
#pragma unroll
  for (int i = ty; i < 32; i += 8)
    tile[i][tx] = W[(size_t)(by + i) * Dd + bx + tx];
  __syncthreads();
#pragma unroll
  for (int i = ty; i < 32; i += 8)
    WT[(size_t)(bx + i) * Dd + by + tx] = f2bf(tile[tx][i]);
}

// ---------------------------------------------------------------------------
// bf16 WMMA GEMM: C[M,N] = A[M,K] * BT[N,K]^T + bias
// Each wave: 32x32 tile (2x2 WMMA tiles), register double-buffered K loop.
// Block: 8 waves = 2(M) x 4(N) -> 64x128 block tile.
// mode 0: bf16 out [B,H,S,hd] ; mode 2: bf16 out [B,H,hd,S] ; mode 3: f32 [M,N]
DEV void store_tile(v8f acc, int mrow, int ncol, const float* bias, void* out,
                    int N, int mode, int S, int H, int lane) {
  const int n    = ncol + (lane & 15);
  const int half = (lane >> 4) & 1;
  const float bi = bias ? bias[n] : 0.0f;
#pragma unroll
  for (int r = 0; r < 8; ++r) {
    const int m    = mrow + r + 8 * half;      // C layout: row r / r+8 by half
    const float vl = acc[r] + bi;
    if (mode == 3) {
      ((float*)out)[(size_t)m * N + n] = vl;
    } else {
      const int b = m / S, s = m % S;
      const int h = n >> 6, d = n & 63;
      const unsigned short bf = f2bf(vl);
      if (mode == 2)
        ((unsigned short*)out)[(((size_t)b * H + h) * 64 + d) * S + s] = bf;
      else
        ((unsigned short*)out)[(((size_t)b * H + h) * S + s) * 64 + d] = bf;
    }
  }
}

__global__ __launch_bounds__(256)
void gemm_bf16_wmma(const unsigned short* __restrict__ A,
                    const unsigned short* __restrict__ BT,
                    const float* __restrict__ bias,
                    void* __restrict__ out,
                    int M, int N, int K, int mode, int S, int H) {
  const int lane = threadIdx.x & 31;
  const int wid  = threadIdx.x >> 5;
  const int mrow = blockIdx.x * 64  + (wid & 1) * 32;
  const int ncol = blockIdx.y * 128 + (wid >> 1) * 32;

  const unsigned short* A0 = A  + (size_t)mrow * K;
  const unsigned short* A1 = A0 + (size_t)16 * K;
  const unsigned short* B0 = BT + (size_t)ncol * K;
  const unsigned short* B1 = B0 + (size_t)16 * K;

  v8f acc00 = {0.f,0.f,0.f,0.f,0.f,0.f,0.f,0.f};
  v8f acc01 = acc00, acc10 = acc00, acc11 = acc00;

  v16bf a0 = load_a_frag(A0, K, 0, lane);
  v16bf a1 = load_a_frag(A1, K, 0, lane);
  v16bf b0 = load_b_frag(B0, K, 0, lane);
  v16bf b1 = load_b_frag(B1, K, 0, lane);

  for (int k = 32; k < K; k += 32) {
    // issue next chunk's loads before consuming current -> loads overlap WMMA
    v16bf na0 = load_a_frag(A0, K, k, lane);
    v16bf na1 = load_a_frag(A1, K, k, lane);
    v16bf nb0 = load_b_frag(B0, K, k, lane);
    v16bf nb1 = load_b_frag(B1, K, k, lane);
    __builtin_prefetch(A0 + (size_t)(lane & 15) * K + k + 32, 0, 1);
    __builtin_prefetch(B0 + (size_t)(lane & 15) * K + k + 32, 0, 1);
    acc00 = wmma_bf16(a0, b0, acc00);
    acc01 = wmma_bf16(a0, b1, acc01);
    acc10 = wmma_bf16(a1, b0, acc10);
    acc11 = wmma_bf16(a1, b1, acc11);
    a0 = na0; a1 = na1; b0 = nb0; b1 = nb1;
  }
  acc00 = wmma_bf16(a0, b0, acc00);
  acc01 = wmma_bf16(a0, b1, acc01);
  acc10 = wmma_bf16(a1, b0, acc10);
  acc11 = wmma_bf16(a1, b1, acc11);

  store_tile(acc00, mrow,      ncol,      bias, out, N, mode, S, H, lane);
  store_tile(acc01, mrow,      ncol + 16, bias, out, N, mode, S, H, lane);
  store_tile(acc10, mrow + 16, ncol,      bias, out, N, mode, S, H, lane);
  store_tile(acc11, mrow + 16, ncol + 16, bias, out, N, mode, S, H, lane);
}

// ---------------------------------------------------------------------------
// Flash attention: 1 wave per 16-query tile, 4 waves/block, online softmax over
// 32-key chunks. K/V tiles for chunk i+1 are staged global->LDS with
// GLOBAL_LOAD_ASYNC_TO_LDS_B128 (ping-pong buffers, ASYNCcnt) while chunk i is
// processed; B-fragments then come from LDS (ds_load).
//
// Dynamic LDS layout per wave (shorts):
//   [   0,  512)  pbuf      16x32 bf16   (P transpose buffer)
//   [ 512, 2560)  kstage[0] 32x64 bf16
//   [2560, 4608)  kstage[1]
//   [4608, 6656)  vstage[0] 64x32 bf16
//   [6656, 8704)  vstage[1]
// -> 17408 bytes/wave, 69632 bytes/block.
#define AT_WAVE_SHORTS 8704
#define AT_WAVE_BYTES  17408

DEV void stage_kv(const unsigned short* __restrict__ Kh,
                  const unsigned short* __restrict__ Vh,
                  int kc, int S, unsigned kOff, unsigned vOff, int lane) {
  {  // K tile: rows kc..kc+31, 64 bf16 each (128B) -> LDS stride 128B
    const int row = lane >> 3;           // 4 rows per instruction
    const int col = (lane & 7) * 8;      // shorts
#pragma unroll
    for (int i = 0; i < 8; ++i) {
      const int r = i * 4 + row;
      async_ld_b128(kOff + (unsigned)(r * 128 + col * 2),
                    Kh + (size_t)(kc + r) * 64 + col);
    }
  }
  {  // V tile: rows d=0..63 of Vt (stride S), 32 bf16 at column kc -> LDS 64B rows
    const int row = lane >> 2;           // 8 rows per instruction
    const int col = (lane & 3) * 8;      // shorts
#pragma unroll
    for (int i = 0; i < 8; ++i) {
      const int d = i * 8 + row;
      async_ld_b128(vOff + (unsigned)(d * 64 + col * 2),
                    Vh + (size_t)d * S + kc + col);
    }
  }
}

__global__ __launch_bounds__(128)
void attn_wmma(const unsigned short* __restrict__ Q,
               const unsigned short* __restrict__ Kk,
               const unsigned short* __restrict__ Vt,
               unsigned short* __restrict__ ctx,
               int Bn, int Hn, int S) {
  extern __shared__ unsigned short smem[];
  const int lane   = threadIdx.x & 31;
  const int wid    = threadIdx.x >> 5;
  const int tile   = blockIdx.x * 4 + wid;
  const int qtiles = S >> 4;
  const int bh = tile / qtiles;
  const int qt = tile % qtiles;
  const int b  = bh / Hn, h = bh % Hn;

  const unsigned short* Qh = Q  + ((size_t)b * Hn + h) * S * 64;
  const unsigned short* Kh = Kk + ((size_t)b * Hn + h) * S * 64;
  const unsigned short* Vh = Vt + ((size_t)b * Hn + h) * 64 * S;

  unsigned short* wbase = smem + (size_t)wid * AT_WAVE_SHORTS;
  unsigned short* pbuf  = wbase;
  unsigned short* kst[2] = { wbase + 512,  wbase + 2560 };
  unsigned short* vst[2] = { wbase + 4608, wbase + 6656 };
  // LDS byte addresses for the async-DMA destination (dynamic LDS starts at
  // groupstaticsize; this kernel has no static LDS).
  const unsigned wOff = __builtin_amdgcn_groupstaticsize() +
                        (unsigned)wid * AT_WAVE_BYTES;
  const unsigned kOff[2] = { wOff + 512u * 2,  wOff + 2560u * 2 };
  const unsigned vOff[2] = { wOff + 4608u * 2, wOff + 6656u * 2 };

  const v16bf qa0 = load_a_frag(Qh + (size_t)(qt * 16) * 64, 64, 0,  lane);
  const v16bf qa1 = load_a_frag(Qh + (size_t)(qt * 16) * 64, 64, 32, lane);

  float m_i[8], l_i[8];
  v8f o[4];
#pragma unroll
  for (int r = 0; r < 8; ++r) { m_i[r] = -3.0e38f; l_i[r] = 0.0f; }
#pragma unroll
  for (int nt = 0; nt < 4; ++nt)
#pragma unroll
    for (int r = 0; r < 8; ++r) o[nt][r] = 0.0f;

  const float scale = 0.125f;            // 1/sqrt(64)

  stage_kv(Kh, Vh, 0, S, kOff[0], vOff[0], lane);     // prologue: chunk 0

  int buf = 0;
  for (int kc = 0; kc < S; kc += 32, buf ^= 1) {
    if (kc + 32 < S) {
      stage_kv(Kh, Vh, kc + 32, S, kOff[buf ^ 1], vOff[buf ^ 1], lane);
      WAIT_ASYNCCNT(0x10);   // allow the 16 just-issued ops to stay in flight
    } else {
      WAIT_ASYNCCNT(0x0);
    }
    const unsigned short* kt = kst[buf];
    const unsigned short* vt = vst[buf];

    // ---- logits: two 16-key tiles x (hd=64 = 2 K32 steps), frags from LDS
    v16bf kb00 = load_b_frag(kt,            64, 0,  lane);
    v16bf kb01 = load_b_frag(kt,            64, 32, lane);
    v16bf kb10 = load_b_frag(kt + 16 * 64,  64, 0,  lane);
    v16bf kb11 = load_b_frag(kt + 16 * 64,  64, 32, lane);
    v8f c0 = {0.f,0.f,0.f,0.f,0.f,0.f,0.f,0.f};
    v8f c1 = c0;
    c0 = wmma_bf16(qa0, kb00, c0);  c0 = wmma_bf16(qa1, kb01, c0);
    c1 = wmma_bf16(qa0, kb10, c1);  c1 = wmma_bf16(qa1, kb11, c1);

    // ---- online softmax: row reductions across 16-lane halves
    float fac[8];
#pragma unroll
    for (int r = 0; r < 8; ++r) {
      const float x0 = c0[r] * scale, x1 = c1[r] * scale;
      float t = fmaxf(x0, x1);
      t = fmaxf(t, __shfl_xor(t, 1, 32));
      t = fmaxf(t, __shfl_xor(t, 2, 32));
      t = fmaxf(t, __shfl_xor(t, 4, 32));
      t = fmaxf(t, __shfl_xor(t, 8, 32));
      const float nm = fmaxf(m_i[r], t);
      fac[r] = __expf(m_i[r] - nm);
      m_i[r] = nm;
      const float p0 = __expf(x0 - nm);
      const float p1 = __expf(x1 - nm);
      float rs = p0 + p1;
      rs += __shfl_xor(rs, 1, 32);
      rs += __shfl_xor(rs, 2, 32);
      rs += __shfl_xor(rs, 4, 32);
      rs += __shfl_xor(rs, 8, 32);
      l_i[r] = l_i[r] * fac[r] + rs;
      c0[r] = p0;
      c1[r] = p1;
    }
#pragma unroll
    for (int nt = 0; nt < 4; ++nt)
#pragma unroll
      for (int r = 0; r < 8; ++r) o[nt][r] *= fac[r];

    // ---- transpose P (C layout) -> A fragment via LDS
    __syncthreads();
    {
      const int nn = lane & 15, half = lane >> 4;
#pragma unroll
      for (int r = 0; r < 8; ++r) {
        const int m = r + 8 * half;
        pbuf[m * 32 + nn]      = f2bf(c0[r]);
        pbuf[m * 32 + 16 + nn] = f2bf(c1[r]);
      }
    }
    __syncthreads();
    FragBF pf;
    {
      const int mm = lane & 15, half = lane >> 4;
      const unsigned int* prow = (const unsigned int*)(pbuf + mm * 32);
#pragma unroll
      for (int r = 0; r < 4; ++r) {
        pf.u[r]     = prow[4 * half + r];
        pf.u[4 + r] = prow[8 + 4 * half + r];
      }
    }

    // ---- PV: 4 hd-column tiles, V fragments from the LDS stage (ldb = 32)
#pragma unroll
    for (int nt = 0; nt < 4; ++nt) {
      v16bf vb = load_b_frag(vt + (size_t)(nt * 16) * 32, 32, 0, lane);
      o[nt] = wmma_bf16(pf.v, vb, o[nt]);
    }
  }

  // ---- normalize, emit ctx rows (A-side layout for the output projection)
  const int nn = lane & 15, half = lane >> 4;
#pragma unroll
  for (int r = 0; r < 8; ++r) {
    const float inv = 1.0f / l_i[r];
    const int s = qt * 16 + r + 8 * half;
    unsigned short* orow = ctx + ((size_t)b * S + s) * (size_t)(Hn * 64) + h * 64;
#pragma unroll
    for (int nt = 0; nt < 4; ++nt)
      orow[nt * 16 + nn] = f2bf(o[nt][r] * inv);
  }
}

// ---------------------------------------------------------------------------
extern "C" void kernel_launch(void* const* d_in, const int* in_sizes, int n_in,
                              void* d_out, int out_size, void* d_ws, size_t ws_size,
                              hipStream_t stream) {
  const int B = 4, S = 2048, D = 1024, H = 16;
  const int M = B * S;  // 8192 tokens

  const float* x  = (const float*)d_in[0];
  const float* Wq = (const float*)d_in[1];
  const float* bq = (const float*)d_in[2];
  const float* Wk = (const float*)d_in[3];
  const float* bk = (const float*)d_in[4];
  const float* Wv = (const float*)d_in[5];
  const float* bv = (const float*)d_in[6];
  const float* Wo = (const float*)d_in[7];
  const float* bo = (const float*)d_in[8];
  float* out = (float*)d_out;

  char* ws = (char*)d_ws;
  const size_t MB = 1ull << 20;
  unsigned short* xb  = (unsigned short*)(ws +  0 * MB);  // 16 MB
  unsigned short* WTq = (unsigned short*)(ws + 16 * MB);  //  2 MB
  unsigned short* WTk = (unsigned short*)(ws + 18 * MB);
  unsigned short* WTv = (unsigned short*)(ws + 20 * MB);
  unsigned short* WTo = (unsigned short*)(ws + 22 * MB);
  unsigned short* Qb  = (unsigned short*)(ws + 24 * MB);  // 16 MB [B,H,S,64]
  unsigned short* Kb  = (unsigned short*)(ws + 40 * MB);  // 16 MB [B,H,S,64]
  unsigned short* Vt  = (unsigned short*)(ws + 56 * MB);  // 16 MB [B,H,64,S]
  unsigned short* ctx = (unsigned short*)(ws + 72 * MB);  // 16 MB [B*S, D]

  cvt_f32_bf16<<<(unsigned)((size_t)M * D / 256), 256, 0, stream>>>(x, xb, (size_t)M * D);
  dim3 tg(D / 32, D / 32);
  cvt_transpose_bf16<<<tg, 256, 0, stream>>>(Wq, WTq, D);
  cvt_transpose_bf16<<<tg, 256, 0, stream>>>(Wk, WTk, D);
  cvt_transpose_bf16<<<tg, 256, 0, stream>>>(Wv, WTv, D);
  cvt_transpose_bf16<<<tg, 256, 0, stream>>>(Wo, WTo, D);

  dim3 gg(M / 64, D / 128);
  gemm_bf16_wmma<<<gg, 256, 0, stream>>>(xb, WTq, bq, Qb, M, D, D, 0, S, H);
  gemm_bf16_wmma<<<gg, 256, 0, stream>>>(xb, WTk, bk, Kb, M, D, D, 0, S, H);
  gemm_bf16_wmma<<<gg, 256, 0, stream>>>(xb, WTv, bv, Vt, M, D, D, 2, S, H);

  attn_wmma<<<(B * H * (S / 16)) / 4, 128, 4 * AT_WAVE_BYTES, stream>>>(
      Qb, Kb, Vt, ctx, B, H, S);

  gemm_bf16_wmma<<<gg, 256, 0, stream>>>(ctx, WTo, bo, out, M, D, D, 3, S, H);
}
